// EpsilonTheta_13245679140902
// MI455X (gfx1250) — matrix operand order repaired and checked by reading, not compiled, not run on hardware
//
#include <hip/hip_runtime.h>
#include <hip/hip_bf16.h>
#include <math.h>

typedef __attribute__((ext_vector_type(16))) _Float16 v16h;
typedef __attribute__((ext_vector_type(8)))  float    v8f;

#define NBATCH 4096
#define TLEN   256
#define TP     260     // T + 4 (circular pad 2 each side)
#define XS     264     // padded LDS row stride
#define NCH    8
#define NL     8

__device__ __forceinline__ float leaky(float v){ return v >= 0.f ? v : 0.4f*v; }
__device__ __forceinline__ float frcp (float v){ return __builtin_amdgcn_rcpf(v); }
__device__ __forceinline__ float fsigm(float v){ return frcp(1.f + __expf(-v)); }
__device__ __forceinline__ float ftanh(float v){
    float e = __expf(-2.f*fabsf(v));
    float t = (1.f - e) * frcp(1.f + e);
    return copysignf(t, v);
}

__global__ __launch_bounds__(32) void eps_theta_kernel(
    const float* __restrict__ inputs, const int* __restrict__ timei,
    const float* __restrict__ cond,
    const float* __restrict__ in_w,  const float* __restrict__ in_b,
    const float* __restrict__ de_w1, const float* __restrict__ de_b1,
    const float* __restrict__ de_w2, const float* __restrict__ de_b2,
    const float* __restrict__ cu_w1, const float* __restrict__ cu_b1,
    const float* __restrict__ cu_w2, const float* __restrict__ cu_b2,
    const float* __restrict__ dc_w,  const float* __restrict__ dc_b,
    const float* __restrict__ dp_w,  const float* __restrict__ dp_b,
    const float* __restrict__ cp_w,  const float* __restrict__ cp_b,
    const float* __restrict__ op_w,  const float* __restrict__ op_b,
    const float* __restrict__ sk_w,  const float* __restrict__ sk_b,
    const float* __restrict__ out_w, const float* __restrict__ out_b,
    float* __restrict__ out)
{
    __shared__ float sXA[NCH*XS];     // activation double buffer A
    __shared__ float sXB[NCH*XS];     // activation double buffer B
    __shared__ float sSkip[NCH*XS];   // skip accumulator
    __shared__ float sCondP[TP];      // circular-padded upsampled cond
    __shared__ float sInp[TLEN];
    __shared__ float sCond[100];
    __shared__ float sC1[128];
    __shared__ float sEmb[32];
    __shared__ float sD1[64], sD2[64];
    __shared__ float sDiff[8], sBias2[16], sCpw[16], sOpb[16];
    __shared__ float sY[16*32];       // dc-conv output, 2 tiles wide
    __shared__ float sZ[8*32];        // gated activation, 2 tiles wide
    __shared__ float sSKW[192], sOUTW[24], sSKB[8];

    const int lane = threadIdx.x;     // wave32: 0..31
    const int b    = blockIdx.x;
    const int m    = lane & 15;
    const int hi   = lane >> 4;

    // ---------------- Stage A: diffusion embedding ----------------
    {
        float ts   = (float)timei[b];
        float freq = powf(10.f, (float)(lane & 15) * 0.25f);  // 10^(j*4/16)
        float arg  = ts * freq;
        sEmb[lane] = (lane < 16) ? sinf(arg) : cosf(arg);
    }
    __syncthreads();
    {   // Linear(32->64) + SiLU : lane computes outputs lane, lane+32
        float a0 = de_b1[lane], a1 = de_b1[lane+32];
        for (int j = 0; j < 32; ++j) {
            float e = sEmb[j];
            a0 += e * de_w1[lane*32 + j];
            a1 += e * de_w1[(lane+32)*32 + j];
        }
        sD1[lane]    = a0 * fsigm(a0);
        sD1[lane+32] = a1 * fsigm(a1);
    }
    __syncthreads();
    {   // Linear(64->64) + SiLU
        float a0 = de_b2[lane], a1 = de_b2[lane+32];
        for (int j = 0; j < 64; ++j) {
            float e = sD1[j];
            a0 += e * de_w2[lane*64 + j];
            a1 += e * de_w2[(lane+32)*64 + j];
        }
        sD2[lane]    = a0 * fsigm(a0);
        sD2[lane+32] = a1 * fsigm(a1);
    }

    // ---------------- Stage B: cond upsampler ----------------
    for (int j = lane; j < 100; j += 32) sCond[j] = cond[b*100 + j];
    __syncthreads();
    for (int q = 0; q < 4; ++q) {     // Linear(100->128) + leaky
        int o = lane + 32*q;
        float acc = cu_b1[o];
        const float* wr = cu_w1 + o*100;
        for (int j = 0; j < 100; ++j) acc += sCond[j] * wr[j];
        sC1[o] = leaky(acc);
    }
    __syncthreads();
    for (int q = 0; q < 8; ++q) {     // Linear(128->256) + leaky, scatter into circular pad
        int o = lane + 32*q;
        float acc = cu_b2[o];
        const float* wr = cu_w2 + o*128;
        for (int j = 0; j < 128; ++j) acc += sC1[j] * wr[j];
        float v = leaky(acc);
        sCondP[o + 2] = v;                       // t = o+2
        if (o < 2)    sCondP[o + 258] = v;       // wrap high end
        if (o >= 254) sCondP[o - 254] = v;       // wrap low end
    }

    // ---------------- Stage C: input projection ----------------
    for (int j = lane; j < TLEN; j += 32) sInp[j] = inputs[b*TLEN + j];
    __syncthreads();
    for (int c = 0; c < NCH; ++c) {
        float wc = in_w[c], bc = in_b[c];
        for (int t = lane; t < TP; t += 32) {
            int ti = (t + 254) & 255;            // (t-2) mod 256
            sXA[c*XS + t]   = leaky(wc * sInp[ti] + bc);
            sSkip[c*XS + t] = 0.f;
        }
    }
    __syncthreads();

    // ---------------- Residual layers (WMMA) ----------------
    float* xin  = sXA;
    float* xout = sXB;
    for (int li = 0; li < NL; ++li) {
        const int dil = 1 << (li & 1);
        // diff[c] = d @ dp_w[li]^T + dp_b
        if (lane < 8) {
            float acc = dp_b[li*8 + lane];
            const float* wr = dp_w + (li*8 + lane)*64;
            for (int h = 0; h < 64; ++h) acc += sD2[h] * wr[h];
            sDiff[lane] = acc;
        }
        __syncthreads();
        // fold diff + cp_b into bias: bias2 = dc_b + sum_ci (sum_k w)*diff + cp_b
        if (lane < 16) {
            float acc = dc_b[li*16 + lane] + cp_b[li*16 + lane];
            const float* wr = dc_w + (li*16 + lane)*24;
            for (int ci = 0; ci < 8; ++ci)
                acc += (wr[ci*3] + wr[ci*3+1] + wr[ci*3+2]) * sDiff[ci];
            sBias2[lane] = acc;
            sCpw[lane] = cp_w[li*16 + lane];
            sOpb[lane] = op_b[li*16 + lane];
        }
        __syncthreads();

        // A fragments (16x32 f16): lane m = lane&15; element e -> K = hi*16+e
        v16h aDC, aOP;
        #pragma unroll
        for (int e = 0; e < 16; ++e) {
            const int  K    = (hi << 4) + e;
            const bool useD = (e < 8) || (hi == 0);        // K < 24
            const bool useO = (e < 8) && (hi == 0);        // K < 8
            float w  = dc_w[(li*16 + m)*24 + (useD ? K : 0)];
            float w2 = op_w[(li*16 + m)*8  + (K & 7)];
            aDC[e] = (_Float16)(useD ? w  : 0.f);
            aOP[e] = (_Float16)(useO ? w2 : 0.f);
        }

        // 18 tiles (last is fully clamped/unstored), processed in pairs for ILP
        for (int tp2 = 0; tp2 < 9; ++tp2) {
            const int tile0 = tp2 * 2;
            const int t0 = tile0*16 + m;
            const int t1 = t0 + 16;
            const int tc0 = (t0 < TP) ? t0 : (TP - 1);
            const int tc1 = (t1 < TP) ? t1 : (TP - 1);

            // B fragments (32x16 f16): lane n = m; element e -> K = hi*16+e
            v16h bF0, bF1;
            #pragma unroll
            for (int e = 0; e < 16; ++e) {
                const bool use = (e < 8) || (hi == 0);     // K = hi*16+e < 24
                const int  ciA = e/3,                kkA = e - (e/3)*3;       // hi==0
                const int  KB  = (e < 8) ? (16+e) : 0;
                const int  ciB = KB/3,               kkB = KB - (KB/3)*3;     // hi==1
                const int  ci  = hi ? ciB : ciA;
                const int  kk  = hi ? kkB : kkA;
                int d0 = tc0 + (kk - 1)*dil;               // circular over [0,260)
                d0 = (d0 < 0) ? d0 + TP : ((d0 >= TP) ? d0 - TP : d0);
                int d1 = tc1 + (kk - 1)*dil;
                d1 = (d1 < 0) ? d1 + TP : ((d1 >= TP) ? d1 - TP : d1);
                float v0 = xin[ci*XS + d0];
                float v1 = xin[ci*XS + d1];
                bF0[e] = (_Float16)(use ? v0 : 0.f);
                bF1[e] = (_Float16)(use ? v1 : 0.f);
            }
            v8f y0 = {};
            y0 = __builtin_amdgcn_wmma_f32_16x16x32_f16(false, aDC, false, bF0, (short)0, y0, false, false);
            v8f y1 = {};
            y1 = __builtin_amdgcn_wmma_f32_16x16x32_f16(false, aDC, false, bF1, (short)0, y1, false, false);
            #pragma unroll
            for (int r = 0; r < 8; ++r) {
                sY[(r + hi*8)*32 + m]      = y0[r];
                sY[(r + hi*8)*32 + 16 + m] = y1[r];
            }
            __syncthreads();

            // gated activation: z = sigmoid(gate)*tanh(filt); col = lane spans both tiles
            {
                const int tq  = tile0*16 + lane;
                const int tqc = (tq < TP) ? tq : (TP - 1);
                const float cpv = sCondP[tqc];
                #pragma unroll
                for (int q = 0; q < 8; ++q) {
                    float g = sY[q*32 + lane]     + sBias2[q]   + sCpw[q]  *cpv;
                    float f = sY[(q+8)*32 + lane] + sBias2[q+8] + sCpw[q+8]*cpv;
                    sZ[q*32 + lane] = fsigm(g) * ftanh(f);
                }
            }
            __syncthreads();

            // output projection (k=1, 8->16) via WMMA, K padded to 32
            v16h bZ0, bZ1;
            #pragma unroll
            for (int e = 0; e < 16; ++e) {
                const bool use = (e < 8) && (hi == 0);     // K = hi*16+e < 8
                float v0 = sZ[(e & 7)*32 + m];
                float v1 = sZ[(e & 7)*32 + 16 + m];
                bZ0[e] = (_Float16)(use ? v0 : 0.f);
                bZ1[e] = (_Float16)(use ? v1 : 0.f);
            }
            v8f o0 = {};
            o0 = __builtin_amdgcn_wmma_f32_16x16x32_f16(false, aOP, false, bZ0, (short)0, o0, false, false);
            v8f o1 = {};
            o1 = __builtin_amdgcn_wmma_f32_16x16x32_f16(false, aOP, false, bZ1, (short)0, o1, false, false);

            #pragma unroll
            for (int r = 0; r < 8; ++r) {
                const int mo = r + hi*8;
                const float ob = sOpb[mo];
                const float v0 = leaky(o0[r] + ob);
                const float v1 = leaky(o1[r] + ob);
                if (hi == 0) {   // residual rows 0..7
                    if (t0 < TP) xout[mo*XS + t0] = (xin[mo*XS + t0] + v0) * 0.70710678118654752f;
                    if (t1 < TP) xout[mo*XS + t1] = (xin[mo*XS + t1] + v1) * 0.70710678118654752f;
                } else {         // skip rows 8..15
                    if (t0 < TP) sSkip[(mo-8)*XS + t0] += v0;
                    if (t1 < TP) sSkip[(mo-8)*XS + t1] += v1;
                }
            }
            __syncthreads();
        }
        { float* tp = xin; xin = xout; xout = tp; }
        __syncthreads();
    }

    // ---------------- Output head ----------------
    for (int j = lane; j < 192; j += 32) sSKW[j] = sk_w[j] * 0.35355339059327f; // fold 1/sqrt(8)
    if (lane < 24) sOUTW[lane] = out_w[lane];
    if (lane < 8)  sSKB[lane]  = sk_b[lane];
    __syncthreads();

    float* sU = xout;   // reuse free x buffer
    for (int c = 0; c < NCH; ++c) {
        for (int t = lane; t < 258; t += 32) {
            float accv = sSKB[c];
            #pragma unroll
            for (int ci = 0; ci < 8; ++ci) {
                accv += sSKW[c*24 + ci*3 + 0] * sSkip[ci*XS + t + 0];
                accv += sSKW[c*24 + ci*3 + 1] * sSkip[ci*XS + t + 1];
                accv += sSKW[c*24 + ci*3 + 2] * sSkip[ci*XS + t + 2];
            }
            sU[c*XS + t] = leaky(accv);
        }
    }
    __syncthreads();

    const float ob = out_b[0];
    for (int t = lane; t < TLEN; t += 32) {
        float accv = ob;
        #pragma unroll
        for (int c = 0; c < 8; ++c) {
            accv += sOUTW[c*3 + 0] * sU[c*XS + t + 0];
            accv += sOUTW[c*3 + 1] * sU[c*XS + t + 1];
            accv += sOUTW[c*3 + 2] * sU[c*XS + t + 2];
        }
        out[b*TLEN + t] = accv;
    }
}

extern "C" void kernel_launch(void* const* d_in, const int* in_sizes, int n_in,
                              void* d_out, int out_size, void* d_ws, size_t ws_size,
                              hipStream_t stream) {
    (void)in_sizes; (void)n_in; (void)out_size; (void)d_ws; (void)ws_size;
    const float* inputs = (const float*)d_in[0];
    const int*   timei  = (const int*)  d_in[1];
    const float* cond   = (const float*)d_in[2];
    const float* in_w   = (const float*)d_in[3];
    const float* in_b   = (const float*)d_in[4];
    const float* de_w1  = (const float*)d_in[5];
    const float* de_b1  = (const float*)d_in[6];
    const float* de_w2  = (const float*)d_in[7];
    const float* de_b2  = (const float*)d_in[8];
    const float* cu_w1  = (const float*)d_in[9];
    const float* cu_b1  = (const float*)d_in[10];
    const float* cu_w2  = (const float*)d_in[11];
    const float* cu_b2  = (const float*)d_in[12];
    const float* dc_w   = (const float*)d_in[13];
    const float* dc_b   = (const float*)d_in[14];
    const float* dp_w   = (const float*)d_in[15];
    const float* dp_b   = (const float*)d_in[16];
    const float* cp_w   = (const float*)d_in[17];
    const float* cp_b   = (const float*)d_in[18];
    const float* op_w   = (const float*)d_in[19];
    const float* op_b   = (const float*)d_in[20];
    const float* sk_w   = (const float*)d_in[21];
    const float* sk_b   = (const float*)d_in[22];
    const float* out_w  = (const float*)d_in[23];
    const float* out_b  = (const float*)d_in[24];

    eps_theta_kernel<<<NBATCH, 32, 0, stream>>>(
        inputs, timei, cond, in_w, in_b, de_w1, de_b1, de_w2, de_b2,
        cu_w1, cu_b1, cu_w2, cu_b2, dc_w, dc_b, dp_w, dp_b,
        cp_w, cp_b, op_w, op_b, sk_w, sk_b, out_w, out_b,
        (float*)d_out);
}
